// DSSA_76347338654279
// MI455X (gfx1250) — matrix operand order repaired and checked by reading, not compiled, not run on hardware
//
#include <hip/hip_runtime.h>
#include <hip/hip_bf16.h>

// ---------------------------------------------------------------------------
// GAT forward for MI455X (gfx1250, wave32).
// GEMMs: v_wmma_f32_16x16x32_f16, 128-row stripes, B transposed in LDS so
// both A and B fragments are K-contiguous (ds_load_b128, no u16 packing).
// Edge stage: float4 gathers + fp32 global atomics (~130MB set fits 192MB L2).
// ---------------------------------------------------------------------------

typedef __attribute__((ext_vector_type(16))) _Float16 v16h;
typedef __attribute__((ext_vector_type(8)))  float    v8f;

#define BSTRIDE 136   // padded halfs per row of transposed B (272B = 17*16B)

// order-preserving float <-> uint key (for atomicMax-based segment max)
__device__ __forceinline__ unsigned f2key(float f) {
    unsigned u = __float_as_uint(f);
    return (u & 0x80000000u) ? ~u : (u | 0x80000000u);
}
__device__ __forceinline__ float key2f(unsigned k) {
    unsigned u = (k & 0x80000000u) ? (k ^ 0x80000000u) : ~k;
    return __uint_as_float(u);
}

__device__ __forceinline__ void edge_sd(const int* __restrict__ ei, int E, int e,
                                        int& s, int& d) {
    if (e < E) { s = ei[e]; d = ei[E + e]; }
    else       { s = e - E; d = e - E; }       // self loops appended
}

// ---------------------------------------------------------------------------
// C[M,128] = A[M,128] @ B[128,128] (+bias). Block = 256 thr = 8 waves,
// computes a 128-row stripe. Wave w owns cols [16w,16w+16) and loops the 8
// row-tiles, reusing its B fragment -> 32 WMMAs/wave, 2 b128 LDS loads (A) each.
// Fragment layouts per CDNA5 ISA 7.12.2 (16-bit A 16x32, B 32x16, f32 C/D).
// ---------------------------------------------------------------------------
__global__ void gemm128_wmma(const float* __restrict__ A,
                             const float* __restrict__ B,
                             const float* __restrict__ bias,
                             float* __restrict__ C, int M)
{
    __shared__ _Float16 sA[128 * 128];        // 32KB  A stripe, row-major
    __shared__ _Float16 sBt[128 * BSTRIDE];   // ~34KB B transposed [n][k]
    const int tid  = threadIdx.x;
    const int row0 = blockIdx.x * 128;

    for (int i = tid; i < 128 * 128; i += 256) {
        int r = i >> 7, c = i & 127;
        int gr = row0 + r;
        sA[i] = (_Float16)((gr < M) ? A[(size_t)gr * 128 + c] : 0.0f);
    }
    for (int i = tid; i < 128 * 128; i += 256) {
        int k = i >> 7, n = i & 127;           // global read coalesced
        sBt[n * BSTRIDE + k] = (_Float16)B[i]; // padded stride spreads banks
    }
    __syncthreads();

    const int wave = tid >> 5;
    const int lane = tid & 31;
    const int col0 = wave * 16;
    const int mrow = lane & 15;               // A row / B col for this lane
    const int kA   = (lane < 16) ? 0 : 8;     // A: upper half-wave starts K=8
    const int kB   = (lane < 16) ? 0 : 8;     // B: upper half-wave starts K=8

    v8f acc[8] = {};
    const _Float16* bBase = &sBt[(col0 + mrow) * BSTRIDE];

    #pragma unroll
    for (int kk = 0; kk < 128; kk += 32) {
        v16h b;                                // b[e] <-> same K order as a
        #pragma unroll
        for (int j = 0; j < 16; ++j)
            b[j] = bBase[kk + ((j < 8) ? (kB + j) : (16 + kB + j - 8))];
        #pragma unroll
        for (int rt = 0; rt < 8; ++rt) {
            const _Float16* aBase = &sA[(rt * 16 + mrow) * 128 + kk];
            v16h a;                            // a[2i(+1)] <-> K-contig pairs
            #pragma unroll
            for (int i = 0; i < 8; ++i) {
                int k0 = (i < 4) ? (kA + 2 * i) : (16 + kA + 2 * (i - 4));
                a[2 * i]     = aBase[k0];
                a[2 * i + 1] = aBase[k0 + 1];
            }
            acc[rt] = __builtin_amdgcn_wmma_f32_16x16x32_f16(
                          false, a, false, b, (short)0, acc[rt], false, false);
        }
    }

    const int rAdd = (lane < 16) ? 0 : 8;     // C/D: vgpr r -> M = r (+8 hi)
    #pragma unroll
    for (int rt = 0; rt < 8; ++rt) {
        #pragma unroll
        for (int r = 0; r < 8; ++r) {
            int gr = row0 + rt * 16 + r + rAdd;
            if (gr < M) {
                int gc = col0 + mrow;
                float val = acc[rt][r];
                if (bias) val += bias[gc];
                C[(size_t)gr * 128 + gc] = val;
            }
        }
    }
}

// h[n,c] += type_emb[node_types[n], c]
__global__ void add_type_emb(float* __restrict__ h, const int* __restrict__ types,
                             const float* __restrict__ temb, int N)
{
    int i = blockIdx.x * blockDim.x + threadIdx.x;
    if (i >= N * 128) return;
    int n = i >> 7, c = i & 127;
    h[i] += temb[types[n] * 128 + c];
}

// s[n,h]=dot(xh[n,h,:],a_src[h,:]) ; d likewise
__global__ void compute_sd(const float* __restrict__ xh,
                           const float* __restrict__ asrc,
                           const float* __restrict__ adst,
                           float* __restrict__ s, float* __restrict__ d, int N)
{
    int i = blockIdx.x * blockDim.x + threadIdx.x;
    if (i >= N * 8) return;
    int n = i >> 3, hd = i & 7;
    const float* row = xh + (size_t)n * 128 + hd * 16;
    const float* as  = asrc + hd * 16;
    const float* ad  = adst + hd * 16;
    float ss = 0.f, dd = 0.f;
    #pragma unroll
    for (int j = 0; j < 16; ++j) { ss += row[j] * as[j]; dd += row[j] * ad[j]; }
    s[i] = ss; d[i] = dd;
}

// pass1: e = leaky_relu(s[src]+d[dst]); store; segment max via atomicMax key
__global__ void edge_max(const int* __restrict__ ei, int E, int Etot,
                         const float* __restrict__ s, const float* __restrict__ d,
                         float* __restrict__ eex, unsigned* __restrict__ mkey)
{
    int i = blockIdx.x * blockDim.x + threadIdx.x;
    if (i >= Etot * 8) return;
    int e = i >> 3, hd = i & 7;
    int sn, dn; edge_sd(ei, E, e, sn, dn);
    float v = s[sn * 8 + hd] + d[dn * 8 + hd];
    v = (v > 0.f) ? v : 0.2f * v;              // leaky_relu 0.2
    eex[i] = v;
    atomicMax(&mkey[dn * 8 + hd], f2key(v));
}

// pass2: ex = exp(e - m[dst]); store; den[dst] += ex
__global__ void edge_expsum(const int* __restrict__ ei, int E, int Etot,
                            const unsigned* __restrict__ mkey,
                            float* __restrict__ eex, float* __restrict__ den)
{
    int i = blockIdx.x * blockDim.x + threadIdx.x;
    if (i >= Etot * 8) return;
    int e = i >> 3, hd = i & 7;
    int sn, dn; edge_sd(ei, E, e, sn, dn);
    float ex = __expf(eex[i] - key2f(mkey[dn * 8 + hd]));
    eex[i] = ex;
    atomicAdd(&den[dn * 8 + hd], ex);
}

// pass3: agg[dst] += xh[src] * alpha. 32 lanes/edge, float4 per lane.
__global__ void edge_scatter(const int* __restrict__ ei, int E, int Etot,
                             const float* __restrict__ xh,
                             const float* __restrict__ eex,
                             const float* __restrict__ den,
                             float* __restrict__ agg)
{
    int i = blockIdx.x * blockDim.x + threadIdx.x;
    if (i >= Etot * 32) return;
    int e = i >> 5, l4 = i & 31;
    int c0 = l4 * 4, hd = l4 >> 2;
    int sn, dn; edge_sd(ei, E, e, sn, dn);
    float alpha = eex[e * 8 + hd] / (den[dn * 8 + hd] + 1e-16f);
    const float4 v = ((const float4*)(xh + (size_t)sn * 128))[l4];
    float* out = agg + (size_t)dn * 128 + c0;
    atomicAdd(out + 0, v.x * alpha);
    atomicAdd(out + 1, v.y * alpha);
    atomicAdd(out + 2, v.z * alpha);
    atomicAdd(out + 3, v.w * alpha);
}

// per node (one wave32): t=agg+b; LayerNorm; h = relu(LN + h_prev)
__global__ void node_ln_relu(const float* __restrict__ agg,
                             const float* __restrict__ bgat,
                             const float* __restrict__ lng,
                             const float* __restrict__ lnb,
                             const float* __restrict__ hprev,
                             float* __restrict__ hout, int N)
{
    int wave = threadIdx.x >> 5, lane = threadIdx.x & 31;
    int n = blockIdx.x * 8 + wave;
    if (n >= N) return;
    int c0 = lane * 4;
    float x[4], s = 0.f, s2 = 0.f;
    #pragma unroll
    for (int j = 0; j < 4; ++j) {
        x[j] = agg[(size_t)n * 128 + c0 + j] + bgat[c0 + j];
        s += x[j]; s2 += x[j] * x[j];
    }
    #pragma unroll
    for (int off = 16; off >= 1; off >>= 1) {
        s  += __shfl_xor(s, off, 32);
        s2 += __shfl_xor(s2, off, 32);
    }
    float mean = s * (1.f / 128.f);
    float var  = s2 * (1.f / 128.f) - mean * mean;
    float inv  = rsqrtf(var + 1e-5f);
    #pragma unroll
    for (int j = 0; j < 4; ++j) {
        float y = (x[j] - mean) * inv * lng[c0 + j] + lnb[c0 + j]
                + hprev[(size_t)n * 128 + c0 + j];
        hout[(size_t)n * 128 + c0 + j] = fmaxf(y, 0.f);
    }
}

// logit[n] = sum_c tanh(t[n,c])*Wa2[c] + ba2 ; track global max (key atomic)
__global__ void attn_logits(const float* __restrict__ t,
                            const float* __restrict__ Wa2,
                            const float* __restrict__ ba2,
                            float* __restrict__ logit,
                            unsigned* __restrict__ maxkey, int N)
{
    int wave = threadIdx.x >> 5, lane = threadIdx.x & 31;
    int n = blockIdx.x * 8 + wave;
    if (n >= N) return;
    int c0 = lane * 4;
    float acc = 0.f;
    #pragma unroll
    for (int j = 0; j < 4; ++j)
        acc += tanhf(t[(size_t)n * 128 + c0 + j]) * Wa2[c0 + j];
    #pragma unroll
    for (int off = 16; off >= 1; off >>= 1) acc += __shfl_xor(acc, off, 32);
    if (lane == 0) {
        float l = acc + ba2[0];
        logit[n] = l;
        atomicMax(maxkey, f2key(l));
    }
}

__global__ void exp_w(const float* __restrict__ logit,
                      const unsigned* __restrict__ maxkey,
                      float* __restrict__ w, float* __restrict__ dens, int N)
{
    int n = blockIdx.x * blockDim.x + threadIdx.x;
    if (n >= N) return;
    float e = __expf(logit[n] - key2f(*maxkey));
    w[n] = e;
    atomicAdd(dens, e);
}

// hg[c] = sum_n h[n,c]*(w[n]/den + 1/N)   (softmax pool + mean pool fused)
__global__ void pool(const float* __restrict__ h, const float* __restrict__ w,
                     const float* __restrict__ dens, float* __restrict__ hg, int N)
{
    int c = threadIdx.x;                       // 128 threads, coalesced rows
    float invden = 1.f / (*dens);
    float invN   = 1.f / (float)N;
    float acc = 0.f;
    for (int n = blockIdx.x; n < N; n += gridDim.x)
        acc += h[(size_t)n * 128 + c] * (w[n] * invden + invN);
    atomicAdd(&hg[c], acc);
}

__global__ void final_mlp(const float* __restrict__ hg,
                          const float* __restrict__ Wc1, const float* __restrict__ bc1,
                          const float* __restrict__ Wc2, const float* __restrict__ bc2,
                          float* __restrict__ out)
{
    __shared__ float sh[128];
    __shared__ float red[128];
    int c = threadIdx.x;
    sh[c] = hg[c];
    __syncthreads();
    float acc = bc1[c];
    for (int k = 0; k < 128; ++k) acc += sh[k] * Wc1[k * 128 + c];
    red[c] = fmaxf(acc, 0.f) * Wc2[c];
    __syncthreads();
    for (int off = 64; off >= 1; off >>= 1) {
        if (c < off) red[c] += red[c + off];
        __syncthreads();
    }
    if (c == 0) out[0] = red[0] + bc2[0];
}

// ---------------------------------------------------------------------------
extern "C" void kernel_launch(void* const* d_in, const int* in_sizes, int n_in,
                              void* d_out, int out_size, void* d_ws, size_t ws_size,
                              hipStream_t stream)
{
    const float* x      = (const float*)d_in[0];
    const int*   ei     = (const int*)  d_in[1];
    const int*   ntypes = (const int*)  d_in[2];
    const float* W_emb  = (const float*)d_in[3];
    const float* b_emb  = (const float*)d_in[4];
    const float* t_emb  = (const float*)d_in[5];
    const float* W_gat  = (const float*)d_in[6];
    const float* a_src  = (const float*)d_in[7];
    const float* a_dst  = (const float*)d_in[8];
    const float* b_gat  = (const float*)d_in[9];
    const float* ln_g   = (const float*)d_in[10];
    const float* ln_b   = (const float*)d_in[11];
    const float* Wa1    = (const float*)d_in[12];
    const float* ba1    = (const float*)d_in[13];
    const float* Wa2    = (const float*)d_in[14];
    const float* ba2    = (const float*)d_in[15];
    const float* Wc1    = (const float*)d_in[16];
    const float* bc1    = (const float*)d_in[17];
    const float* Wc2    = (const float*)d_in[18];
    const float* bc2    = (const float*)d_in[19];

    const int N    = in_sizes[2];
    const int E    = in_sizes[1] / 2;
    const int Etot = E + N;

    char* p = (char*)d_ws;
    auto alloc = [&](size_t bytes) {
        void* r = (void*)p;
        p += (bytes + 255) & ~(size_t)255;
        return r;
    };
    float*    hA     = (float*)   alloc((size_t)N * 128 * 4);
    float*    hB     = (float*)   alloc((size_t)N * 128 * 4);
    float*    xh     = (float*)   alloc((size_t)N * 128 * 4);
    float*    agg    = (float*)   alloc((size_t)N * 128 * 4);
    float*    sarr   = (float*)   alloc((size_t)N * 8 * 4);
    float*    darr   = (float*)   alloc((size_t)N * 8 * 4);
    unsigned* mkey   = (unsigned*)alloc((size_t)N * 8 * 4);
    float*    den8   = (float*)   alloc((size_t)N * 8 * 4);
    float*    eex    = (float*)   alloc((size_t)Etot * 8 * 4);
    float*    logit  = (float*)   alloc((size_t)N * 4);
    float*    wbuf   = (float*)   alloc((size_t)N * 4);
    float*    hg     = (float*)   alloc(128 * 4);
    unsigned* maxkey = (unsigned*)alloc(4);
    float*    dens   = (float*)   alloc(4);
    (void)ws_size; (void)n_in; (void)out_size;

    const int gB = (N + 127) / 128;                 // GEMM blocks (128-row stripes)
    const int nW = (N + 7) / 8;                     // wave-per-node blocks

    // embedding: h = x@W_emb + b_emb + type_emb[types]
    gemm128_wmma<<<gB, 256, 0, stream>>>(x, W_emb, b_emb, hA, N);
    add_type_emb<<<(N * 128 + 255) / 256, 256, 0, stream>>>(hA, ntypes, t_emb, N);

    float* hcur = hA; float* hnext = hB;
    for (int l = 0; l < 3; ++l) {
        gemm128_wmma<<<gB, 256, 0, stream>>>(hcur, W_gat + (size_t)l * 128 * 128,
                                             nullptr, xh, N);
        compute_sd<<<(N * 8 + 255) / 256, 256, 0, stream>>>(
            xh, a_src + l * 128, a_dst + l * 128, sarr, darr, N);

        hipMemsetAsync(mkey, 0, (size_t)N * 8 * 4, stream);   // key 0 == -NaN (min)
        hipMemsetAsync(den8, 0, (size_t)N * 8 * 4, stream);
        hipMemsetAsync(agg,  0, (size_t)N * 128 * 4, stream);

        edge_max<<<(Etot * 8 + 255) / 256, 256, 0, stream>>>(ei, E, Etot, sarr, darr,
                                                             eex, mkey);
        edge_expsum<<<(Etot * 8 + 255) / 256, 256, 0, stream>>>(ei, E, Etot, mkey,
                                                                eex, den8);
        edge_scatter<<<(Etot * 32 + 255) / 256, 256, 0, stream>>>(ei, E, Etot, xh,
                                                                  eex, den8, agg);
        node_ln_relu<<<nW, 256, 0, stream>>>(agg, b_gat + l * 128, ln_g + l * 128,
                                             ln_b + l * 128, hcur, hnext, N);
        float* t = hcur; hcur = hnext; hnext = t;
    }

    // pooling head
    gemm128_wmma<<<gB, 256, 0, stream>>>(hcur, Wa1, ba1, xh, N);
    hipMemsetAsync(maxkey, 0, 4, stream);
    hipMemsetAsync(dens,   0, 4, stream);
    hipMemsetAsync(hg,     0, 128 * 4, stream);
    attn_logits<<<nW, 256, 0, stream>>>(xh, Wa2, ba2, logit, maxkey, N);
    exp_w<<<(N + 255) / 256, 256, 0, stream>>>(logit, maxkey, wbuf, dens, N);
    pool<<<1024, 128, 0, stream>>>(hcur, wbuf, dens, hg, N);
    final_mlp<<<1, 128, 0, stream>>>(hg, Wc1, bc1, Wc2, bc2, (float*)d_out);
}
